// TransformerBlock_50483045597391
// MI455X (gfx1250) — matrix operand order repaired and checked by reading, not compiled, not run on hardware
//
#include <hip/hip_runtime.h>
#include <hip/hip_bf16.h>

// ---------------------------------------------------------------------------
// MI455X transformer block. All matmuls via v_wmma_f32_16x16x32_bf16 (wave32).
// ~155 GFLOP matmul vs ~100MB traffic @23.3TB/s -> matrix-core bound. Each wave
// computes a 16x32 tile (one A fragment feeds 2-4 WMMAs). K-loop is unrolled by
// two with the body's loads grouped ahead of its WMMAs; amdgpu_waves_per_eu(1)
// widens the register budget so the scheduler can double-buffer fragments
// instead of WAR-reusing one buffer (which forced s_wait_loadcnt 0x0).
// ---------------------------------------------------------------------------

#define D_MODEL 1024
#define NUM_HEADS 16
#define D_FF 4096
#define BATCH 4
#define SEQ 1024
#define DK 64
#define NTOK (BATCH * SEQ)

#define GEMM_ATTRS __launch_bounds__(256) __attribute__((amdgpu_waves_per_eu(1)))

typedef __bf16 bf16_t;
typedef __attribute__((ext_vector_type(16))) __bf16 v16bf_t;
typedef __attribute__((ext_vector_type(8)))  __bf16 v8bf_t;
typedef __attribute__((ext_vector_type(8)))  float  v8f_t;

#define ZERO8 v8f_t{0.f, 0.f, 0.f, 0.f, 0.f, 0.f, 0.f, 0.f}

__device__ __forceinline__ bf16_t f2bf(float f) {
  union { float f; unsigned u; } in;
  in.f = f;
  unsigned u = in.u;
  u += 0x7fffu + ((u >> 16) & 1u);            // round to nearest even
  union { unsigned short s; bf16_t b; } out;
  out.s = (unsigned short)(u >> 16);
  return out.b;
}

// A-operand fragment (16x32 bf16, MxK). Lane l holds row M=l&15.
// lanes 0-15: K=0..7 (vgpr0-3), K=16..23 (vgpr4-7); lanes 16-31: K=8..15, 24..31.
__device__ __forceinline__ v16bf_t load_a_frag(const bf16_t* row, int kb, int lane) {
  union { v16bf_t v; v8bf_t h[2]; } u;
  int half = (lane & 16) >> 1;                // 0 or 8 elements
  u.h[0] = *(const v8bf_t*)(row + kb + half);
  u.h[1] = *(const v8bf_t*)(row + kb + 16 + half);
  return u.v;
}

// B-operand fragment (32x16 bf16, KxN). Lane l holds column N=l&15.
// lanes 0-15: K=0..7, 16..23; lanes 16-31: K=8..15, 24..31 (same as A).
__device__ __forceinline__ v16bf_t load_b_frag(const bf16_t* row, int kb, int lane) {
  union { v16bf_t v; v8bf_t h[2]; } u;
  int half = (lane & 16) >> 1;
  u.h[0] = *(const v8bf_t*)(row + kb + half);
  u.h[1] = *(const v8bf_t*)(row + kb + 16 + half);
  return u.v;
}

__device__ __forceinline__ v8f_t wmma_bf16(v16bf_t a, v16bf_t b, v8f_t c) {
  return __builtin_amdgcn_wmma_f32_16x16x32_bf16(false, a, false, b, (short)0, c,
                                                 false, false);
}

// One A fragment feeds two WMMAs. Unrolled x2: both half-steps' loads are
// issued before the first WMMA so matrix math overlaps outstanding vmem.
__device__ __forceinline__ void gemm_loop2(const bf16_t* __restrict__ arow,
                                           const bf16_t* __restrict__ b0row,
                                           const bf16_t* __restrict__ b1row,
                                           int Kd, int lane, v8f_t& c0, v8f_t& c1) {
  for (int kb = 0; kb < Kd; kb += 64) {
    __builtin_prefetch(arow + kb + 576, 0, 1);   // global_prefetch_b8
    __builtin_prefetch(b0row + kb + 576, 0, 1);
    __builtin_prefetch(b1row + kb + 576, 0, 1);
    v16bf_t a0  = load_a_frag(arow, kb, lane);
    v16bf_t b00 = load_b_frag(b0row, kb, lane);
    v16bf_t b10 = load_b_frag(b1row, kb, lane);
    v16bf_t a1  = load_a_frag(arow, kb + 32, lane);
    v16bf_t b01 = load_b_frag(b0row, kb + 32, lane);
    v16bf_t b11 = load_b_frag(b1row, kb + 32, lane);
    c0 = wmma_bf16(a0, b00, c0);
    c1 = wmma_bf16(a0, b10, c1);
    c0 = wmma_bf16(a1, b01, c0);
    c1 = wmma_bf16(a1, b11, c1);
  }
}

// ---------------------------------------------------------------------------

__global__ void __launch_bounds__(256)
k_f32_to_bf16(const float* __restrict__ in, bf16_t* __restrict__ out, int n) {
  int i = blockIdx.x * blockDim.x + threadIdx.x;
  if (i < n) out[i] = f2bf(in[i]);
}

__global__ void __launch_bounds__(256)
k_rmsnorm(const float* __restrict__ x, const float* __restrict__ g,
          bf16_t* __restrict__ y) {
  int row = blockIdx.x;
  const float* xr = x + (size_t)row * D_MODEL;
  float ss = 0.f;
  for (int i = threadIdx.x; i < D_MODEL; i += 256) { float v = xr[i]; ss += v * v; }
  #pragma unroll
  for (int off = 16; off >= 1; off >>= 1) ss += __shfl_xor(ss, off, 32);
  __shared__ float red[8];
  if ((threadIdx.x & 31) == 0) red[threadIdx.x >> 5] = ss;
  __syncthreads();
  float tot = 0.f;
  #pragma unroll
  for (int wv = 0; wv < 8; ++wv) tot += red[wv];
  float rinv = rsqrtf(tot * (1.0f / D_MODEL) + 1e-5f);
  bf16_t* yr = y + (size_t)row * D_MODEL;
  for (int i = threadIdx.x; i < D_MODEL; i += 256) yr[i] = f2bf(xr[i] * g[i] * rinv);
}

// RoPE + store for one 16x16 C tile of a Q/K projection.
__device__ __forceinline__ void rope_store(v8f_t c, int col, int rbase, int lane,
                                           bf16_t* __restrict__ out) {
  int h = col >> 6, dk = col & 63;
  // inv_freq = 10000^(-(dk&~1)/64) = exp(-(dk>>1)*ln(1e4)/32)
  float invf = __expf(-(float)(dk >> 1) * 0.28782313662425572f);
  #pragma unroll
  for (int i = 0; i < 8; ++i) {
    int tok = rbase + i;
    int b = tok >> 10, s = tok & (SEQ - 1);
    float sn, cs;
    __sincosf((float)s * invf, &sn, &cs);
    float v = c[i];
    float p = __shfl_xor(v, 1, 32);   // partner column lives in adjacent lane
    float r = (col & 1) ? (p * sn + v * cs) : (v * cs - p * sn);
    out[((size_t)(b * NUM_HEADS + h) * SEQ + s) * DK + dk] = f2bf(r);
  }
}

// QKV projection: block tile 64x64 (8 waves of 16x32). Fused RoPE epilogue for
// Q/K. Q,K out: (B,H,S,DK). V out transposed: (B,H,DK,S) for attention B-frags.
__global__ void GEMM_ATTRS
k_qkv(const bf16_t* __restrict__ y, const bf16_t* __restrict__ wq,
      const bf16_t* __restrict__ wk, const bf16_t* __restrict__ wv,
      bf16_t* __restrict__ qo, bf16_t* __restrict__ ko, bf16_t* __restrict__ vto) {
  int lane = threadIdx.x & 31;
  int wid = threadIdx.x >> 5;
  int m0 = blockIdx.x * 64 + (wid >> 1) * 16;
  int n0 = blockIdx.y * 64 + (wid & 1) * 32;
  int mode = blockIdx.z;
  const bf16_t* W = (mode == 0) ? wq : (mode == 1) ? wk : wv;
  v8f_t c0 = ZERO8, c1 = ZERO8;
  gemm_loop2(y + (size_t)(m0 + (lane & 15)) * D_MODEL,
             W + (size_t)(n0 + (lane & 15)) * D_MODEL,
             W + (size_t)(n0 + 16 + (lane & 15)) * D_MODEL, D_MODEL, lane, c0, c1);
  int col = n0 + (lane & 15);
  int rbase = m0 + ((lane >> 4) << 3);
  if (mode < 2) {
    bf16_t* out = (mode == 0) ? qo : ko;
    rope_store(c0, col, rbase, lane, out);
    rope_store(c1, col + 16, rbase, lane, out);
  } else {
    #pragma unroll
    for (int i = 0; i < 8; ++i) {
      int tok = rbase + i;
      int b = tok >> 10, s = tok & (SEQ - 1);
      int h0 = col >> 6, d0 = col & 63;
      int h1 = (col + 16) >> 6, d1 = (col + 16) & 63;
      vto[((size_t)(b * NUM_HEADS + h0) * DK + d0) * SEQ + s] = f2bf(c0[i]);
      vto[((size_t)(b * NUM_HEADS + h1) * DK + d1) * SEQ + s] = f2bf(c1[i]);
    }
  }
}

// Flash attention: one wave per (b,h,16-query tile). Scores via 4 WMMAs per
// 32-key block (K-dim = dk = 64). Online softmax is lane-local in the C layout;
// P goes through a per-wave LDS tile to be re-read in A layout; P@V is 4 WMMAs
// (one P fragment reused) against transposed V.
__global__ void GEMM_ATTRS
k_attn(const bf16_t* __restrict__ q, const bf16_t* __restrict__ k,
       const bf16_t* __restrict__ vt, bf16_t* __restrict__ ctx) {
  __shared__ __align__(64) bf16_t Pl[8][16][32];
  int lane = threadIdx.x & 31;
  int w = threadIdx.x >> 5;
  int wid = blockIdx.x * 8 + w;
  int b = wid >> 10;
  int h = (wid >> 6) & 15;
  int q0 = (wid & 63) << 4;
  size_t bh = (size_t)(b * NUM_HEADS + h);
  const bf16_t* qbase = q + bh * SEQ * DK;
  const bf16_t* kbase = k + bh * SEQ * DK;
  const bf16_t* vbase = vt + bh * DK * SEQ;

  const bf16_t* qrow = qbase + (size_t)(q0 + (lane & 15)) * DK;
  v16bf_t qa0 = load_a_frag(qrow, 0, lane);
  v16bf_t qa1 = load_a_frag(qrow, 32, lane);

  float m[8], l[8];
  v8f_t o0 = ZERO8, o1 = ZERO8, o2 = ZERO8, o3 = ZERO8;
  #pragma unroll
  for (int i = 0; i < 8; ++i) { m[i] = -__builtin_inff(); l[i] = 0.f; }

  int colk = lane & 15;
  int rbase = (lane >> 4) << 3;
  const float scale = 0.125f;  // 1/sqrt(64)

  for (int kb = 0; kb <= q0 + 15; kb += 32) {
    const bf16_t* kr0 = kbase + (size_t)(kb + colk) * DK;
    const bf16_t* kr1 = kbase + (size_t)(kb + 16 + colk) * DK;
    v8f_t s0 = ZERO8, s1 = ZERO8;
    s0 = wmma_bf16(qa0, load_b_frag(kr0, 0, lane), s0);
    s0 = wmma_bf16(qa1, load_b_frag(kr0, 32, lane), s0);
    s1 = wmma_bf16(qa0, load_b_frag(kr1, 0, lane), s1);
    s1 = wmma_bf16(qa1, load_b_frag(kr1, 32, lane), s1);

    #pragma unroll
    for (int i = 0; i < 8; ++i) {
      int qr = q0 + rbase + i;
      float a = (kb + colk <= qr) ? s0[i] * scale : -__builtin_inff();
      float bb = (kb + 16 + colk <= qr) ? s1[i] * scale : -__builtin_inff();
      float rm = fmaxf(a, bb);
      rm = fmaxf(rm, __shfl_xor(rm, 1, 32));
      rm = fmaxf(rm, __shfl_xor(rm, 2, 32));
      rm = fmaxf(rm, __shfl_xor(rm, 4, 32));
      rm = fmaxf(rm, __shfl_xor(rm, 8, 32));
      float mn = fmaxf(m[i], rm);
      float alpha = __expf(m[i] - mn);
      m[i] = mn;
      float e0 = __expf(a - mn);
      float e1 = __expf(bb - mn);
      float sum = e0 + e1;
      sum += __shfl_xor(sum, 1, 32);
      sum += __shfl_xor(sum, 2, 32);
      sum += __shfl_xor(sum, 4, 32);
      sum += __shfl_xor(sum, 8, 32);
      l[i] = l[i] * alpha + sum;
      o0[i] *= alpha; o1[i] *= alpha; o2[i] *= alpha; o3[i] *= alpha;
      Pl[w][rbase + i][colk] = f2bf(e0);
      Pl[w][rbase + i][16 + colk] = f2bf(e1);
    }
    // P: C-layout -> A-layout via per-wave LDS tile (DS ops in-order per wave)
    v16bf_t pa = load_a_frag(&Pl[w][lane & 15][0], 0, lane);
    o0 = wmma_bf16(pa, load_b_frag(vbase + (size_t)(0 + colk) * SEQ, kb, lane), o0);
    o1 = wmma_bf16(pa, load_b_frag(vbase + (size_t)(16 + colk) * SEQ, kb, lane), o1);
    o2 = wmma_bf16(pa, load_b_frag(vbase + (size_t)(32 + colk) * SEQ, kb, lane), o2);
    o3 = wmma_bf16(pa, load_b_frag(vbase + (size_t)(48 + colk) * SEQ, kb, lane), o3);
  }

  #pragma unroll
  for (int i = 0; i < 8; ++i) {
    float rinv = 1.0f / l[i];
    int s = q0 + rbase + i;
    bf16_t* crow = ctx + ((size_t)(b * SEQ + s) * NUM_HEADS + h) * DK;
    crow[0 + colk]  = f2bf(o0[i] * rinv);
    crow[16 + colk] = f2bf(o1[i] * rinv);
    crow[32 + colk] = f2bf(o2[i] * rinv);
    crow[48 + colk] = f2bf(o3[i] * rinv);
  }
}

__global__ void GEMM_ATTRS
k_wo(const bf16_t* __restrict__ ctx, const bf16_t* __restrict__ wo,
     const float* __restrict__ x, float* __restrict__ x1) {
  int lane = threadIdx.x & 31;
  int wid = threadIdx.x >> 5;
  int m0 = blockIdx.x * 64 + (wid >> 1) * 16;
  int n0 = blockIdx.y * 64 + (wid & 1) * 32;
  v8f_t c0 = ZERO8, c1 = ZERO8;
  gemm_loop2(ctx + (size_t)(m0 + (lane & 15)) * D_MODEL,
             wo + (size_t)(n0 + (lane & 15)) * D_MODEL,
             wo + (size_t)(n0 + 16 + (lane & 15)) * D_MODEL, D_MODEL, lane, c0, c1);
  int col = n0 + (lane & 15);
  int rbase = m0 + ((lane >> 4) << 3);
  #pragma unroll
  for (int i = 0; i < 8; ++i) {
    size_t idx = (size_t)(rbase + i) * D_MODEL + col;
    x1[idx] = x[idx] + c0[i];
    x1[idx + 16] = x[idx + 16] + c1[i];
  }
}

// Fused W1/W3 GEMM + SiLU-GLU epilogue. One A fragment feeds 4 WMMAs
// (two 16-col tiles x two weight matrices), K-loop unrolled x2 with grouped loads.
__global__ void GEMM_ATTRS
k_ffn1(const bf16_t* __restrict__ y, const bf16_t* __restrict__ w1,
       const bf16_t* __restrict__ w3, bf16_t* __restrict__ glu) {
  int lane = threadIdx.x & 31;
  int wid = threadIdx.x >> 5;
  int m0 = blockIdx.x * 64 + (wid >> 1) * 16;
  int n0 = blockIdx.y * 64 + (wid & 1) * 32;
  const bf16_t* arow = y + (size_t)(m0 + (lane & 15)) * D_MODEL;
  const bf16_t* b1a = w1 + (size_t)(n0 + (lane & 15)) * D_MODEL;
  const bf16_t* b1b = w1 + (size_t)(n0 + 16 + (lane & 15)) * D_MODEL;
  const bf16_t* b3a = w3 + (size_t)(n0 + (lane & 15)) * D_MODEL;
  const bf16_t* b3b = w3 + (size_t)(n0 + 16 + (lane & 15)) * D_MODEL;
  v8f_t c1a = ZERO8, c1b = ZERO8, c3a = ZERO8, c3b = ZERO8;
  for (int kb = 0; kb < D_MODEL; kb += 64) {
    __builtin_prefetch(arow + kb + 576, 0, 1);
    __builtin_prefetch(b1a + kb + 576, 0, 1);
    __builtin_prefetch(b3a + kb + 576, 0, 1);
    v16bf_t a0   = load_a_frag(arow, kb, lane);
    v16bf_t f1a0 = load_b_frag(b1a, kb, lane);
    v16bf_t f1b0 = load_b_frag(b1b, kb, lane);
    v16bf_t f3a0 = load_b_frag(b3a, kb, lane);
    v16bf_t f3b0 = load_b_frag(b3b, kb, lane);
    v16bf_t a1   = load_a_frag(arow, kb + 32, lane);
    v16bf_t f1a1 = load_b_frag(b1a, kb + 32, lane);
    v16bf_t f1b1 = load_b_frag(b1b, kb + 32, lane);
    v16bf_t f3a1 = load_b_frag(b3a, kb + 32, lane);
    v16bf_t f3b1 = load_b_frag(b3b, kb + 32, lane);
    c1a = wmma_bf16(a0, f1a0, c1a);
    c1b = wmma_bf16(a0, f1b0, c1b);
    c3a = wmma_bf16(a0, f3a0, c3a);
    c3b = wmma_bf16(a0, f3b0, c3b);
    c1a = wmma_bf16(a1, f1a1, c1a);
    c1b = wmma_bf16(a1, f1b1, c1b);
    c3a = wmma_bf16(a1, f3a1, c3a);
    c3b = wmma_bf16(a1, f3b1, c3b);
  }
  int col = n0 + (lane & 15);
  int rbase = m0 + ((lane >> 4) << 3);
  #pragma unroll
  for (int i = 0; i < 8; ++i) {
    float la = c1a[i], lb = c1b[i];
    float sa = 1.0f / (1.0f + __expf(-la));
    float sb = 1.0f / (1.0f + __expf(-lb));
    size_t idx = (size_t)(rbase + i) * D_FF + col;
    glu[idx] = f2bf(sa * la * c3a[i]);
    glu[idx + 16] = f2bf(sb * lb * c3b[i]);
  }
}

__global__ void GEMM_ATTRS
k_ffn2(const bf16_t* __restrict__ glu, const bf16_t* __restrict__ w2,
       const float* __restrict__ x1, float* __restrict__ out) {
  int lane = threadIdx.x & 31;
  int wid = threadIdx.x >> 5;
  int m0 = blockIdx.x * 64 + (wid >> 1) * 16;
  int n0 = blockIdx.y * 64 + (wid & 1) * 32;
  v8f_t c0 = ZERO8, c1 = ZERO8;
  gemm_loop2(glu + (size_t)(m0 + (lane & 15)) * D_FF,
             w2 + (size_t)(n0 + (lane & 15)) * D_FF,
             w2 + (size_t)(n0 + 16 + (lane & 15)) * D_FF, D_FF, lane, c0, c1);
  int col = n0 + (lane & 15);
  int rbase = m0 + ((lane >> 4) << 3);
  #pragma unroll
  for (int i = 0; i < 8; ++i) {
    size_t idx = (size_t)(rbase + i) * D_MODEL + col;
    out[idx] = x1[idx] + c0[i];
    out[idx + 16] = x1[idx + 16] + c1[i];
  }
}

// ---------------------------------------------------------------------------

extern "C" void kernel_launch(void* const* d_in, const int* in_sizes, int n_in,
                              void* d_out, int out_size, void* d_ws, size_t ws_size,
                              hipStream_t stream) {
  (void)in_sizes; (void)n_in; (void)out_size; (void)ws_size;
  const float* x  = (const float*)d_in[0];
  const float* wq = (const float*)d_in[1];
  const float* wk = (const float*)d_in[2];
  const float* wv = (const float*)d_in[3];
  const float* wo = (const float*)d_in[4];
  const float* w1 = (const float*)d_in[5];
  const float* w2 = (const float*)d_in[6];
  const float* w3 = (const float*)d_in[7];
  const float* g1 = (const float*)d_in[8];
  const float* g2 = (const float*)d_in[9];

  char* ws = (char*)d_ws;
  size_t off = 0;
  auto take = [&](size_t bytes) -> void* {
    void* p = ws + off;
    off += (bytes + 255) & ~(size_t)255;
    return p;
  };
  const size_t DD = (size_t)D_MODEL * D_MODEL;
  const size_t FD = (size_t)D_FF * D_MODEL;
  const size_t TD = (size_t)NTOK * D_MODEL;
  const size_t TF = (size_t)NTOK * D_FF;

  bf16_t* wq_b = (bf16_t*)take(DD * 2);
  bf16_t* wk_b = (bf16_t*)take(DD * 2);
  bf16_t* wv_b = (bf16_t*)take(DD * 2);
  bf16_t* wo_b = (bf16_t*)take(DD * 2);
  bf16_t* w1_b = (bf16_t*)take(FD * 2);
  bf16_t* w3_b = (bf16_t*)take(FD * 2);
  bf16_t* w2_b = (bf16_t*)take(FD * 2);
  bf16_t* y_b  = (bf16_t*)take(TD * 2);   // reused for y1 then y2
  bf16_t* q_b  = (bf16_t*)take(TD * 2);
  bf16_t* k_b  = (bf16_t*)take(TD * 2);
  bf16_t* vt_b = (bf16_t*)take(TD * 2);
  bf16_t* ctx_b = (bf16_t*)take(TD * 2);
  float*  x1   = (float*)take(TD * 4);
  bf16_t* glu_b = (bf16_t*)take(TF * 2);

  auto cvt = [&](const float* src, bf16_t* dst, size_t n) {
    k_f32_to_bf16<<<dim3((unsigned)((n + 255) / 256)), dim3(256), 0, stream>>>(
        src, dst, (int)n);
  };
  cvt(wq, wq_b, DD); cvt(wk, wk_b, DD); cvt(wv, wv_b, DD); cvt(wo, wo_b, DD);
  cvt(w1, w1_b, FD); cvt(w3, w3_b, FD); cvt(w2, w2_b, FD);

  k_rmsnorm<<<dim3(NTOK), dim3(256), 0, stream>>>(x, g1, y_b);
  k_qkv<<<dim3(NTOK / 64, D_MODEL / 64, 3), dim3(256), 0, stream>>>(
      y_b, wq_b, wk_b, wv_b, q_b, k_b, vt_b);
  k_attn<<<dim3(BATCH * NUM_HEADS * (SEQ / 16) / 8), dim3(256), 0, stream>>>(
      q_b, k_b, vt_b, ctx_b);
  k_wo<<<dim3(NTOK / 64, D_MODEL / 64), dim3(256), 0, stream>>>(ctx_b, wo_b, x, x1);
  k_rmsnorm<<<dim3(NTOK), dim3(256), 0, stream>>>(x1, g2, y_b);
  k_ffn1<<<dim3(NTOK / 64, D_FF / 64), dim3(256), 0, stream>>>(y_b, w1_b, w3_b, glu_b);
  k_ffn2<<<dim3(NTOK / 64, D_MODEL / 64), dim3(256), 0, stream>>>(
      glu_b, w2_b, x1, (float*)d_out);
}
